// CausalSelfAttention_18159121728202
// MI455X (gfx1250) — matrix operand order repaired and checked by reading, not compiled, not run on hardware
//
#include <hip/hip_runtime.h>
#include <hip/hip_bf16.h>

#define Bn 4
#define Sn 2048
#define Dn 768
#define Hn 12
#define HDn 64
#define QKV3 (3*Dn)          // 2304
#define Mrows (Bn*Sn)        // 8192

typedef __bf16 bf16x16 __attribute__((ext_vector_type(16)));
typedef float  f32x8   __attribute__((ext_vector_type(8)));

union FragBF { bf16x16 v; unsigned int u[8]; };

__device__ __forceinline__ unsigned short f2bf(float f) {
    union { float f; unsigned int u; } c; c.f = f;
    unsigned int u = c.u;
    unsigned int r = (u + 0x7FFFu + ((u >> 16) & 1u)) >> 16;
    return (unsigned short)r;
}

// ---------------- conversion kernels ----------------
__global__ void cvt_f32_bf16_k(const float* __restrict__ x, unsigned short* __restrict__ y, int n) {
    for (int i = blockIdx.x * blockDim.x + threadIdx.x; i < n; i += gridDim.x * blockDim.x)
        y[i] = f2bf(x[i]);
}

// w: [K][N] fp32 -> wT: [N][K] bf16
__global__ void transpose_cvt_k(const float* __restrict__ w, unsigned short* __restrict__ wT, int K, int N) {
    int total = K * N;
    for (int i = blockIdx.x * blockDim.x + threadIdx.x; i < total; i += gridDim.x * blockDim.x) {
        int n = i / K, k = i - n * K;
        wT[i] = f2bf(w[(size_t)k * N + n]);
    }
}

// ---------------- WMMA GEMM: C[M,N] = A[M,K](bf16) * Bt[N,K]^T(bf16) + bias ----------------
// 128x128 block tile, 256 threads (8 waves), each wave owns 32(M) x 64(N):
// 2 A-frags x 4 B-frags -> 8 WMMAs per K-step with B-fragment reuse across M.
template <bool OUT_BF16>
__global__ __launch_bounds__(256) void gemm_wmma_k(
    const unsigned short* __restrict__ A,   // [M,K] bf16
    const unsigned short* __restrict__ Bt,  // [N,K] bf16 (transposed weight)
    const float* __restrict__ bias,         // [N]
    void* __restrict__ Out,                 // bf16 [M,N] or f32 [M,N]
    int M, int N, int K)
{
    __shared__ unsigned int lA[128 * 16];  // 128 rows x 32 bf16
    __shared__ unsigned int lB[128 * 16];  // 128 rows x 32 bf16

    const int m0 = blockIdx.y * 128;
    const int n0 = blockIdx.x * 128;
    const int t    = threadIdx.x;
    const int wid  = t >> 5;        // 0..7
    const int lane = t & 31;
    const int wm   = wid >> 1;      // 0..3  row subtile (32 rows)
    const int wn   = wid & 1;       // 0..1  col subtile (64 cols)
    const int half = lane >> 4;
    const int l16  = lane & 15;

    f32x8 acc[2][4] = {};

    for (int k0 = 0; k0 < K; k0 += 32) {
        #pragma unroll
        for (int i = 0; i < 2; ++i) {   // stage A and B tiles: 512 uint4 each
            int idx = t + i * 256;
            int row = idx >> 2, seg = idx & 3;
            ((uint4*)lA)[idx] = ((const uint4*)(A  + (size_t)(m0 + row) * K + k0))[seg];
            ((uint4*)lB)[idx] = ((const uint4*)(Bt + (size_t)(n0 + row) * K + k0))[seg];
        }
        __syncthreads();

        FragBF a[2];
        #pragma unroll
        for (int w = 0; w < 2; ++w) {
            int row = wm * 32 + w * 16 + l16;
            #pragma unroll
            for (int v = 0; v < 8; ++v) {
                int kk2 = half * 4 + (v >> 2) * 8 + (v & 3);   // u32 index in row
                a[w].u[v] = lA[row * 16 + kk2];
            }
        }
        #pragma unroll
        for (int f = 0; f < 4; ++f) {
            FragBF bq;
            int nn = wn * 64 + f * 16 + l16;
            #pragma unroll
            for (int v = 0; v < 8; ++v)
                bq.u[v] = lB[nn * 16 + half * 4 + (v >> 2) * 8 + (v & 3)];
            acc[0][f] = __builtin_amdgcn_wmma_f32_16x16x32_bf16(
                false, a[0].v, false, bq.v, (short)0, acc[0][f], false, false);
            acc[1][f] = __builtin_amdgcn_wmma_f32_16x16x32_bf16(
                false, a[1].v, false, bq.v, (short)0, acc[1][f], false, false);
        }
        __syncthreads();
    }

    #pragma unroll
    for (int f = 0; f < 4; ++f) {
        int ncol = n0 + wn * 64 + f * 16 + l16;
        float bv = bias[ncol];
        #pragma unroll
        for (int w = 0; w < 2; ++w) {
            #pragma unroll
            for (int r = 0; r < 8; ++r) {
                int row = m0 + wm * 32 + w * 16 + half * 8 + r;
                float val = acc[w][f][r] + bv;
                if (OUT_BF16)
                    ((unsigned short*)Out)[(size_t)row * N + ncol] = f2bf(val);
                else
                    ((float*)Out)[(size_t)row * N + ncol] = val;
            }
        }
    }
}

// ---------------- flash attention (causal), bf16 WMMA ----------------
__global__ __launch_bounds__(128) void attn_wmma_k(
    const unsigned short* __restrict__ qkv,   // [B*S, 2304] bf16, [q|k|v] per head
    unsigned short* __restrict__ attn_out)    // [B*S, 768] bf16
{
    __shared__ unsigned int lK[64 * 32];             // 64 keys x 64 hd bf16
    __shared__ unsigned int lV[64 * 32];
    __shared__ unsigned short lP[4][16 * 64];        // per-wave P tile

    const int qt = blockIdx.x;   // 0..31  (64 q rows)
    const int h  = blockIdx.y;   // 0..11
    const int b  = blockIdx.z;   // 0..3
    const int t    = threadIdx.x;
    const int wid  = t >> 5;
    const int lane = t & 31;
    const int half = lane >> 4;
    const int l16  = lane & 15;

    const unsigned int* qkv_u = (const unsigned int*)qkv;   // row stride 1152 u32

    // ---- Q fragments (HD=64 -> 2 A-frags of 16x32) ----
    FragBF q0, q1;
    {
        size_t qrow = (size_t)(b * Sn + qt * 64 + wid * 16 + l16) * 1152 + h * 32;
        #pragma unroll
        for (int v = 0; v < 8; ++v) {
            int kk2 = half * 4 + (v >> 2) * 8 + (v & 3);
            q0.u[v] = qkv_u[qrow + kk2];
            q1.u[v] = qkv_u[qrow + 16 + kk2];
        }
    }

    f32x8 acc[4] = {};
    float mrow[8], lrow[8];
    #pragma unroll
    for (int r = 0; r < 8; ++r) { mrow[r] = -3.0e38f; lrow[r] = 0.f; }

    for (int jt = 0; jt <= qt; ++jt) {
        { // cooperative K/V tile load: 2 threads/row, 4 uint4 each
            int row = t >> 1, colh = t & 1;
            size_t gbase = (size_t)(b * Sn + jt * 64 + row) * 1152 + h * 32;
            const uint4* srcK = (const uint4*)(qkv_u + gbase + 384 + colh * 16);
            const uint4* srcV = (const uint4*)(qkv_u + gbase + 768 + colh * 16);
            uint4* dK = (uint4*)lK;
            uint4* dV = (uint4*)lV;
            #pragma unroll
            for (int i = 0; i < 4; ++i) {
                dK[row * 8 + colh * 4 + i] = srcK[i];
                dV[row * 8 + colh * 4 + i] = srcV[i];
            }
        }
        __syncthreads();

        // ---- scores: S = Q @ K^T  (4 frags of 16x16 over 64 keys) ----
        f32x8 s[4];
        #pragma unroll
        for (int f = 0; f < 4; ++f) {
            FragBF bk0, bk1;
            int nn = f * 16 + l16;    // key within tile
            #pragma unroll
            for (int v = 0; v < 8; ++v) {
                bk0.u[v] = lK[nn * 32 + half * 4 + (v >> 2) * 8 + (v & 3)];
                bk1.u[v] = lK[nn * 32 + 16 + half * 4 + (v >> 2) * 8 + (v & 3)];
            }
            f32x8 z = {};
            z = __builtin_amdgcn_wmma_f32_16x16x32_bf16(false, q0.v, false, bk0.v, (short)0, z, false, false);
            z = __builtin_amdgcn_wmma_f32_16x16x32_bf16(false, q1.v, false, bk1.v, (short)0, z, false, false);
            s[f] = z;
        }

        // ---- scale + causal mask ----
        #pragma unroll
        for (int f = 0; f < 4; ++f) {
            int key = jt * 64 + f * 16 + l16;
            #pragma unroll
            for (int r = 0; r < 8; ++r) {
                int qrow = qt * 64 + wid * 16 + half * 8 + r;
                float vv = s[f][r] * 0.125f;
                s[f][r] = (key <= qrow) ? vv : -3.0e38f;
            }
        }

        // ---- online softmax ----
        float tmax[8], tsum[8];
        #pragma unroll
        for (int r = 0; r < 8; ++r) {
            float v = s[0][r];
            v = fmaxf(v, s[1][r]); v = fmaxf(v, s[2][r]); v = fmaxf(v, s[3][r]);
            v = fmaxf(v, __shfl_xor(v, 1, 32));
            v = fmaxf(v, __shfl_xor(v, 2, 32));
            v = fmaxf(v, __shfl_xor(v, 4, 32));
            v = fmaxf(v, __shfl_xor(v, 8, 32));
            tmax[r] = v;
        }
        float mnew[8], scl[8];
        #pragma unroll
        for (int r = 0; r < 8; ++r) {
            mnew[r] = fmaxf(mrow[r], tmax[r]);
            scl[r]  = __expf(mrow[r] - mnew[r]);
            mrow[r] = mnew[r];
        }
        #pragma unroll
        for (int r = 0; r < 8; ++r) {
            float sum = 0.f;
            #pragma unroll
            for (int f = 0; f < 4; ++f) {
                float p = __expf(s[f][r] - mnew[r]);
                s[f][r] = p;
                sum += p;
            }
            sum += __shfl_xor(sum, 1, 32);
            sum += __shfl_xor(sum, 2, 32);
            sum += __shfl_xor(sum, 4, 32);
            sum += __shfl_xor(sum, 8, 32);
            tsum[r] = sum;
        }
        #pragma unroll
        for (int r = 0; r < 8; ++r) lrow[r] = lrow[r] * scl[r] + tsum[r];
        #pragma unroll
        for (int f = 0; f < 4; ++f)
            #pragma unroll
            for (int r = 0; r < 8; ++r) acc[f][r] *= scl[r];

        // ---- P (C layout) -> LDS -> A layout ----
        #pragma unroll
        for (int f = 0; f < 4; ++f)
            #pragma unroll
            for (int r = 0; r < 8; ++r)
                lP[wid][(half * 8 + r) * 64 + f * 16 + l16] = f2bf(s[f][r]);
        asm volatile("s_wait_dscnt 0" ::: "memory");

        FragBF p0, p1;
        {
            const unsigned int* lPu = (const unsigned int*)lP[wid];
            #pragma unroll
            for (int v = 0; v < 8; ++v) {
                int kk2 = half * 4 + (v >> 2) * 8 + (v & 3);
                p0.u[v] = lPu[l16 * 32 + kk2];
                p1.u[v] = lPu[l16 * 32 + 16 + kk2];
            }
        }

        // ---- acc += P @ V ----
        const unsigned short* lV16 = (const unsigned short*)lV;
        #pragma unroll
        for (int g = 0; g < 4; ++g) {
            FragBF bv0, bv1;
            #pragma unroll
            for (int v = 0; v < 8; ++v) {
                int kk = half * 8 + (v >> 2) * 16 + 2 * (v & 3);
                unsigned int lo = lV16[(kk)     * 64 + g * 16 + l16];
                unsigned int hi = lV16[(kk + 1) * 64 + g * 16 + l16];
                bv0.u[v] = lo | (hi << 16);
                lo = lV16[(kk + 32) * 64 + g * 16 + l16];
                hi = lV16[(kk + 33) * 64 + g * 16 + l16];
                bv1.u[v] = lo | (hi << 16);
            }
            acc[g] = __builtin_amdgcn_wmma_f32_16x16x32_bf16(false, p0.v, false, bv0.v, (short)0, acc[g], false, false);
            acc[g] = __builtin_amdgcn_wmma_f32_16x16x32_bf16(false, p1.v, false, bv1.v, (short)0, acc[g], false, false);
        }
        __syncthreads();
    }

    // ---- normalize + store [B*S, 768] bf16 ----
    float inv[8];
    #pragma unroll
    for (int r = 0; r < 8; ++r) inv[r] = 1.0f / lrow[r];
    #pragma unroll
    for (int g = 0; g < 4; ++g) {
        int col = h * 64 + g * 16 + l16;
        #pragma unroll
        for (int r = 0; r < 8; ++r) {
            int row = b * Sn + qt * 64 + wid * 16 + half * 8 + r;
            attn_out[(size_t)row * Dn + col] = f2bf(acc[g][r] * inv[r]);
        }
    }
}

extern "C" void kernel_launch(void* const* d_in, const int* in_sizes, int n_in,
                              void* d_out, int out_size, void* d_ws, size_t ws_size,
                              hipStream_t stream) {
    const float* x      = (const float*)d_in[0];
    const float* w_qkv  = (const float*)d_in[1];
    const float* b_qkv  = (const float*)d_in[2];
    const float* w_proj = (const float*)d_in[3];
    const float* b_proj = (const float*)d_in[4];
    float* out = (float*)d_out;

    char* ws = (char*)d_ws;
    unsigned short* xbf      = (unsigned short*)(ws);                         // 8192*768
    unsigned short* wqkvT    = (unsigned short*)(ws + 12582912);              // 2304*768
    unsigned short* wprojT   = (unsigned short*)(ws + 16121856);              // 768*768
    unsigned short* qkv_bf   = (unsigned short*)(ws + 17301504);              // 8192*2304
    unsigned short* attn_out = (unsigned short*)(ws + 55050240);              // 8192*768

    cvt_f32_bf16_k<<<4096, 256, 0, stream>>>(x, xbf, Mrows * Dn);
    transpose_cvt_k<<<2048, 256, 0, stream>>>(w_qkv, wqkvT, Dn, QKV3);
    transpose_cvt_k<<<1024, 256, 0, stream>>>(w_proj, wprojT, Dn, Dn);

    gemm_wmma_k<true><<<dim3(QKV3 / 128, Mrows / 128), 256, 0, stream>>>(
        xbf, wqkvT, b_qkv, (void*)qkv_bf, Mrows, QKV3, Dn);

    attn_wmma_k<<<dim3(Sn / 64, Hn, Bn), 128, 0, stream>>>(qkv_bf, attn_out);

    gemm_wmma_k<false><<<dim3(Dn / 128, Mrows / 128), 256, 0, stream>>>(
        attn_out, wprojT, b_proj, (void*)out, Mrows, Dn, Dn);
}
